// MultipleProjections_2688649527674
// MI455X (gfx1250) — matrix operand order repaired and checked by reading, not compiled, not run on hardware
//
#include <hip/hip_runtime.h>

typedef __attribute__((ext_vector_type(16))) __bf16 v16bf;
typedef __attribute__((ext_vector_type(8)))  float  v8f;

#define T_TOK 8192
#define DIM   2048
#define OUT   2048
#define NEXP  8
#define BM    64   // tokens per block row tile (2 M-group waves x 32)

// scratch header (int indices)
#define H_COUNT 0   // [8]  per-expert token counts
#define H_CUR   8   // [8]  scatter cursors
#define H_OFF   16  // [9]  exclusive token offsets
#define H_TOFF  25  // [9]  exclusive row-tile offsets
#define H_TOTAL 34  // total row tiles

__device__ __forceinline__ unsigned short f32_bf16_rne(float f) {
  unsigned u = __builtin_bit_cast(unsigned, f);
  u += 0x7FFFu + ((u >> 16) & 1u);
  return (unsigned short)(u >> 16);
}

__global__ void zero_hdr_kernel(int* hdr) {
  if (threadIdx.x < 64) hdr[threadIdx.x] = 0;
}

__global__ void cvt_bf16_kernel(const float* __restrict__ in,
                                unsigned short* __restrict__ out, int n4) {
  int i = blockIdx.x * blockDim.x + threadIdx.x;
  if (i >= n4) return;
  float4 f = ((const float4*)in)[i];
  ushort4 r;
  r.x = f32_bf16_rne(f.x); r.y = f32_bf16_rne(f.y);
  r.z = f32_bf16_rne(f.z); r.w = f32_bf16_rne(f.w);
  ((ushort4*)out)[i] = r;
}

// One wave32 per token: 8 router dot products, Wr staged in LDS.
__global__ void router_kernel(const float* __restrict__ x, const float* __restrict__ Wr,
                              const float* __restrict__ br, int* __restrict__ expert,
                              int* __restrict__ hdr) {
  __shared__ float wS[NEXP * DIM];  // 64 KB
  for (int i = threadIdx.x; i < NEXP * DIM; i += blockDim.x) wS[i] = Wr[i];
  __syncthreads();
  int wave = threadIdx.x >> 5, lane = threadIdx.x & 31;
  int t = blockIdx.x * 8 + wave;
  const float* xr = x + (size_t)t * DIM;
  float acc[NEXP];
#pragma unroll
  for (int e = 0; e < NEXP; e++) acc[e] = 0.f;
  for (int d = lane; d < DIM; d += 32) {
    float xv = xr[d];
#pragma unroll
    for (int e = 0; e < NEXP; e++) acc[e] += xv * wS[e * DIM + d];
  }
#pragma unroll
  for (int e = 0; e < NEXP; e++) {
#pragma unroll
    for (int o = 16; o > 0; o >>= 1) acc[e] += __shfl_xor(acc[e], o, 32);
  }
  if (lane == 0) {
    float best = acc[0] + br[0]; int bi = 0;   // strict > keeps first index (argmax tie rule)
#pragma unroll
    for (int e = 1; e < NEXP; e++) {
      float v = acc[e] + br[e];
      if (v > best) { best = v; bi = e; }
    }
    expert[t] = bi;
    atomicAdd(&hdr[H_COUNT + bi], 1);
  }
}

__global__ void setup_kernel(int* hdr, float* loss_out) {
  if (threadIdx.x != 0) return;
  const float mean = (float)T_TOK / NEXP;
  float s = 0.f;
  int off = 0, toff = 0;
  for (int e = 0; e < NEXP; e++) {
    int c = hdr[H_COUNT + e];
    float v = (float)c - mean;
    s += v * v;
    hdr[H_OFF + e] = off;
    hdr[H_CUR + e] = off;
    hdr[H_TOFF + e] = toff;
    off  += c;
    toff += (c + BM - 1) / BM;
  }
  hdr[H_OFF + NEXP]  = off;
  hdr[H_TOFF + NEXP] = toff;
  hdr[H_TOTAL]       = toff;
  loss_out[0] = (s / NEXP) * 3e-06f;
}

__global__ void scatter_kernel(const int* __restrict__ expert, int* __restrict__ hdr,
                               int* __restrict__ perm) {
  int t = blockIdx.x * blockDim.x + threadIdx.x;
  if (t >= T_TOK) return;
  int e = expert[t];
  int pos = atomicAdd(&hdr[H_CUR + e], 1);
  perm[pos] = t;  // order within expert irrelevant: per-token math identical
}

union AB32 { uint4 u[2]; v16bf v; };

// Grouped GEMM: grid.x = padded 64-token row tiles, grid.y = OUT/256 panels.
// 8 waves/block arranged 2(M) x 4(N) over a 64x256 block tile. Each wave computes
// a 32x64 fp32 tile: 2 A-frags x 4 B-frags, each B fragment feeds two WMMAs from
// registers (1.5 b128 loads per WMMA); the M-pair waves share B lines via WGP$,
// so L2 weight traffic is amortized over 64 tokens.
__global__ void __launch_bounds__(256)
moe_gemm_kernel(const unsigned short* __restrict__ xb, const unsigned short* __restrict__ Wb,
                const float* __restrict__ bp, const int* __restrict__ hdr,
                const int* __restrict__ perm, float* __restrict__ out) {
  int rt = blockIdx.x;
  if (rt >= hdr[H_TOTAL]) return;
  int e = 0;
  while (e < NEXP - 1 && rt >= hdr[H_TOFF + e + 1]) e++;
  int lt  = rt - hdr[H_TOFF + e];
  int cnt = hdr[H_COUNT + e];

  __shared__ int tokS[BM];
  if (threadIdx.x < BM) {
    int r = lt * BM + threadIdx.x;
    tokS[threadIdx.x] = (r < cnt) ? perm[hdr[H_OFF + e] + r] : -1;
  }
  __syncthreads();

  int wave = threadIdx.x >> 5, lane = threadIdx.x & 31;
  int mg = wave & 1, ng = wave >> 1;          // 2(M) x 4(N) wave grid
  int m = lane & 15, kh = lane >> 4;
  int rbase = mg * 32;

  // Two A fragments (rows rbase+m, rbase+m+16); clamp invalid tail rows (stores masked).
  int tok0 = tokS[rbase + m], tok1 = tokS[rbase + m + 16];
  int tl0 = (tok0 >= 0) ? tok0 : tokS[0];
  int tl1 = (tok1 >= 0) ? tok1 : tokS[0];
  const unsigned short* xrow0 = xb + (size_t)tl0 * DIM + kh * 8;   // ISA A layout
  const unsigned short* xrow1 = xb + (size_t)tl1 * DIM + kh * 8;

  int ocol = blockIdx.y * 256 + ng * 64 + m;                       // B/D column = lane&15
  const unsigned short* w[4];
#pragma unroll
  for (int j = 0; j < 4; j++)
    w[j] = Wb + ((size_t)e * OUT + ocol + 16 * j) * DIM + kh * 8;  // ISA B layout

  v8f acc[2][4];
#pragma unroll
  for (int mi = 0; mi < 2; mi++)
#pragma unroll
    for (int j = 0; j < 4; j++) acc[mi][j] = (v8f){};

  for (int kk = 0; kk < DIM; kk += 32) {
    AB32 a0, a1;
    a0.u[0] = *(const uint4*)(xrow0 + kk);
    a0.u[1] = *(const uint4*)(xrow0 + kk + 16);
    a1.u[0] = *(const uint4*)(xrow1 + kk);
    a1.u[1] = *(const uint4*)(xrow1 + kk + 16);
#pragma unroll
    for (int j = 0; j < 4; j++) {
      AB32 b;
      b.u[0] = *(const uint4*)(w[j] + kk);      // k 8h..8h+7, then 16+8h..16+8h+7
      b.u[1] = *(const uint4*)(w[j] + kk + 16);
      acc[0][j] = __builtin_amdgcn_wmma_f32_16x16x32_bf16(
          false, a0.v, false, b.v, (short)0, acc[0][j], false, false);
      acc[1][j] = __builtin_amdgcn_wmma_f32_16x16x32_bf16(
          false, a1.v, false, b.v, (short)0, acc[1][j], false, false);
    }
  }

  float bias[4];
#pragma unroll
  for (int j = 0; j < 4; j++) bias[j] = bp[e * OUT + ocol + 16 * j];

#pragma unroll
  for (int mi = 0; mi < 2; mi++) {
#pragma unroll
    for (int v = 0; v < 8; v++) {
      int mr = rbase + mi * 16 + v + kh * 8;    // ISA D layout: row = vgpr + 8*(lane>>4)
      int t = tokS[mr];
      if (t >= 0) {
#pragma unroll
        for (int j = 0; j < 4; j++)
          out[(size_t)t * OUT + ocol + 16 * j] = acc[mi][j][v] + bias[j];
      }
    }
  }
}

extern "C" void kernel_launch(void* const* d_in, const int* in_sizes, int n_in,
                              void* d_out, int out_size, void* d_ws, size_t ws_size,
                              hipStream_t stream) {
  const float* x  = (const float*)d_in[0];
  const float* Wp = (const float*)d_in[1];
  const float* bp = (const float*)d_in[2];
  const float* Wr = (const float*)d_in[3];
  const float* br = (const float*)d_in[4];
  float* out = (float*)d_out;

  char* ws = (char*)d_ws;
  int* hdr    = (int*)ws;                              // 256 B header
  int* expert = (int*)(ws + 256);                      // T ints
  int* perm   = (int*)(ws + 256 + 4 * T_TOK);          // T ints
  unsigned short* xb = (unsigned short*)(ws + 256 + 8 * T_TOK);                    // 32 MB
  unsigned short* Wb = (unsigned short*)(ws + 256 + 8 * T_TOK +
                                         (size_t)2 * T_TOK * DIM);                 // 64 MB

  zero_hdr_kernel<<<1, 64, 0, stream>>>(hdr);

  int n4x = T_TOK * DIM / 4;
  int n4w = NEXP * OUT * DIM / 4;
  cvt_bf16_kernel<<<(n4x + 255) / 256, 256, 0, stream>>>(x,  xb, n4x);
  cvt_bf16_kernel<<<(n4w + 255) / 256, 256, 0, stream>>>(Wp, Wb, n4w);

  router_kernel<<<T_TOK / 8, 256, 0, stream>>>(x, Wr, br, expert, hdr);
  setup_kernel<<<1, 32, 0, stream>>>(hdr, out + (size_t)T_TOK * OUT);
  scatter_kernel<<<T_TOK / 256, 256, 0, stream>>>(expert, hdr, perm);

  int maxTiles = T_TOK / BM + NEXP - 1;  // 135: worst-case padded 64-token row tiles
  moe_gemm_kernel<<<dim3(maxTiles, OUT / 256), 256, 0, stream>>>(xb, Wb, bp, hdr, perm, out);
}
